// MultiHeadedAttentionBoost_69106023792701
// MI455X (gfx1250) — matrix operand order repaired and checked by reading, not compiled, not run on hardware
//
#include <hip/hip_runtime.h>
#include <hip/hip_bf16.h>

// ---------------------------------------------------------------------------
// MultiHeadedAttentionBoost on MI455X (gfx1250, wave32, WMMA bf16 + TDM)
// B=8, S=2048, D=1024, H=4, DK=256
// ---------------------------------------------------------------------------

typedef __attribute__((ext_vector_type(16))) __bf16 bf16x16;
typedef __attribute__((ext_vector_type(4)))  __bf16 bf16x4;
typedef __attribute__((ext_vector_type(8)))  float  f32x8;
typedef __attribute__((ext_vector_type(4)))  float  f32x4;
typedef __attribute__((ext_vector_type(4)))  unsigned int u32x4;
typedef __attribute__((ext_vector_type(4)))  int    i32x4;
typedef __attribute__((ext_vector_type(8)))  int    i32x8;

#define B_   8
#define S_   2048
#define D_   1024
#define H_   4
#define DK_  256
#define NEG_ (-1.0e9f)

// --- LDS byte offset of a generic pointer (for TDM descriptors) ------------
__device__ __forceinline__ unsigned lds_offset(const void* p) {
  return (unsigned)(unsigned long long)(const __attribute__((address_space(3))) char*)p;
}

// --- Tensor Data Mover: 2D tile load, data_size = 8B units -----------------
// D# packing per CDNA5 ISA (group0) / §8.4 (group1).
__device__ __forceinline__ void tdm_load_2d(unsigned lds, unsigned long long ga,
                                            unsigned tile_d0, unsigned tile_d1,
                                            unsigned long long tensor_d0,
                                            unsigned long long tensor_d1,
                                            unsigned long long stride0) {
  u32x4 g0 = {};
  g0[0] = 1u;                                       // count=1, user D#
  g0[1] = lds;                                      // lds_addr [63:32]
  g0[2] = (unsigned)(ga & 0xffffffffull);           // global_addr lo
  g0[3] = (unsigned)((ga >> 32) & 0x1ffffffull) |   // global_addr hi (57b)
          0x80000000u;                              // type=2 ("image") [127:126]
  i32x8 g1 = {};
  g1[0] = (int)(3u << 16);                          // data_size=3 (8B)
  g1[1] = (int)((tensor_d0 & 0xffffull) << 16);
  g1[2] = (int)(((tensor_d0 >> 16) & 0xffffull) |
                ((tensor_d1 & 0xffffull) << 16));
  g1[3] = (int)(((tensor_d1 >> 16) & 0xffffull) |
                ((unsigned long long)(tile_d0 & 0xffffu) << 16));
  g1[4] = (int)(tile_d1 & 0xffffu);                 // tile_d1 (tile_d2=0)
  g1[5] = (int)(stride0 & 0xffffffffull);
  g1[6] = (int)((stride0 >> 32) & 0xffffull);
  g1[7] = 0;
  i32x4 z4 = {};
#if __clang_major__ >= 23
  i32x8 z8 = {};
  __builtin_amdgcn_tensor_load_to_lds(g0, g1, z4, z4, z8, 0);
#else
  __builtin_amdgcn_tensor_load_to_lds(g0, g1, z4, z4, 0);
#endif
}

// --- WMMA fragment loaders per CDNA5 ISA §7.12.2 (wave32) -------
__device__ __forceinline__ bf16x16 load_a_frag(const __bf16* __restrict__ base,
                                               int ldk, int lane) {
  const int m  = lane & 15;
  const int k0 = (lane & 16) ? 8 : 0;
  const __bf16* p = base + (size_t)m * ldk + k0;
  bf16x16 a;
#pragma unroll
  for (int i = 0; i < 8; ++i) { a[i] = p[i]; a[8 + i] = p[16 + i]; }
  return a;
}

__device__ __forceinline__ bf16x16 load_b_frag(const __bf16* __restrict__ base,
                                               int ldk, int lane) {
  const int n  = lane & 15;
  const int k0 = (lane & 16) ? 8 : 0;
  const __bf16* p = base + (size_t)n * ldk + k0;
  bf16x16 b;
#pragma unroll
  for (int i = 0; i < 8; ++i) { b[i] = p[i]; b[8 + i] = p[16 + i]; }
  return b;
}

__device__ __forceinline__ f32x8 wmma_bf16(bf16x16 a, bf16x16 b, f32x8 c) {
  return __builtin_amdgcn_wmma_f32_16x16x32_bf16(false, a, false, b,
                                                 (short)0, c, false, false);
}

// ---------------------------------------------------------------------------
// GEMM: Out[M,N] = A[M,K] @ W[N,K]^T + bias.  Block tile 128x64, 8 waves,
// double-buffered LDS, ILP-4 WMMA.  bf16 A-tiles staged by TDM.
// MODE 0: bf16 [b,h,s,dk] (Q,K)   MODE 1: bf16 [b,h,dk,s] (V, transposed)
// MODE 2: f32 [m,n] (output projection)
// ---------------------------------------------------------------------------
template <int MODE, bool ABF16>
__global__ __launch_bounds__(256) void gemm_wmma_kernel(
    const void* __restrict__ Av, const float* __restrict__ W,
    const float* __restrict__ bias, void* __restrict__ Outv,
    int M, int N, int K) {
  __shared__ __bf16 As[2][128 * 32];
  __shared__ __bf16 Bs[2][64 * 32];

  const int tid  = threadIdx.x;
  const int lane = tid & 31;
  const int wid  = tid >> 5;
  const int wm   = wid & 3;
  const int wn   = wid >> 2;
  const int m0   = blockIdx.x * 128;
  const int n0   = blockIdx.y * 64;

  const float*  Af = (const float*)Av;
  const __bf16* Ab = (const __bf16*)Av;

  auto stageA = [&](int k0, int buf) {
#pragma unroll
    for (int i = 0; i < 4; ++i) {
      int v = tid + i * 256;
      int r = v >> 3, c = (v & 7) * 4;
      f32x4 f = *(const f32x4*)&Af[(size_t)(m0 + r) * K + k0 + c];
      bf16x4 o;
#pragma unroll
      for (int j = 0; j < 4; ++j) o[j] = (__bf16)f[j];
      *(bf16x4*)&As[buf][r * 32 + c] = o;
    }
  };
  auto stageA_tdm = [&](int k0, int buf) {  // 2D TDM: 128 rows x 64B, stride 2KB
    tdm_load_2d(lds_offset(&As[buf][0]),
                (unsigned long long)(size_t)&Ab[(size_t)m0 * K + k0],
                /*tile_d0=*/8, /*tile_d1=*/128,
                (unsigned long long)K / 4, (unsigned long long)M,
                (unsigned long long)K / 4);
  };
  auto stageB = [&](int k0, int buf) {
#pragma unroll
    for (int i = 0; i < 2; ++i) {
      int v = tid + i * 256;
      int r = v >> 3, c = (v & 7) * 4;
      f32x4 f = *(const f32x4*)&W[(size_t)(n0 + r) * K + k0 + c];
      bf16x4 o;
#pragma unroll
      for (int j = 0; j < 4; ++j) o[j] = (__bf16)f[j];
      *(bf16x4*)&Bs[buf][r * 32 + c] = o;
    }
  };

  const f32x8 vzero = {};
  f32x8 acc[2][2];
#pragma unroll
  for (int i = 0; i < 2; ++i)
#pragma unroll
    for (int j = 0; j < 2; ++j) acc[i][j] = vzero;

  if (ABF16) { if (wid == 0) stageA_tdm(0, 0); }
  else        stageA(0, 0);
  stageB(0, 0);
  if (ABF16 && wid == 0) __builtin_amdgcn_s_wait_tensorcnt(0);

  const int steps = K / 32;
  for (int s = 0; s < steps; ++s) {
    __syncthreads();
    const int cur = s & 1, nxt = cur ^ 1;
    if (s + 1 < steps) {
      const int k0n = (s + 1) * 32;
      if (ABF16) { if (wid == 0) stageA_tdm(k0n, nxt); }
      else        stageA(k0n, nxt);
      stageB(k0n, nxt);
      if (s + 2 < steps)
        __builtin_prefetch((const char*)&W[(size_t)(n0 + (tid & 63)) * K + k0n + 32], 0, 1);
    }

    bf16x16 aF[2], bF[2];
#pragma unroll
    for (int f = 0; f < 2; ++f)
      aF[f] = load_a_frag(&As[cur][(wm * 32 + f * 16) * 32], 32, lane);
#pragma unroll
    for (int f = 0; f < 2; ++f)
      bF[f] = load_b_frag(&Bs[cur][(wn * 32 + f * 16) * 32], 32, lane);
#pragma unroll
    for (int i = 0; i < 2; ++i)
#pragma unroll
      for (int j = 0; j < 2; ++j) acc[i][j] = wmma_bf16(aF[i], bF[j], acc[i][j]);

    if (ABF16 && wid == 0 && s + 1 < steps)
      __builtin_amdgcn_s_wait_tensorcnt(0);
  }

  const int rsel = (lane >> 4) << 3;
  const int nl   = lane & 15;
#pragma unroll
  for (int i = 0; i < 2; ++i)
#pragma unroll
    for (int j = 0; j < 2; ++j)
#pragma unroll
      for (int r = 0; r < 8; ++r) {
        int mg = m0 + wm * 32 + i * 16 + r + rsel;
        int ng = n0 + wn * 32 + j * 16 + nl;
        float v = acc[i][j][r] + bias[ng];
        if (MODE == 0) {
          int b = mg >> 11, s2 = mg & (S_ - 1);
          int h = ng >> 8,  dk = ng & (DK_ - 1);
          ((__bf16*)Outv)[(((size_t)b * H_ + h) * S_ + s2) * DK_ + dk] = (__bf16)v;
        } else if (MODE == 1) {
          int b = mg >> 11, s2 = mg & (S_ - 1);
          int h = ng >> 8,  dk = ng & (DK_ - 1);
          ((__bf16*)Outv)[(((size_t)b * H_ + h) * DK_ + dk) * S_ + s2] = (__bf16)v;
        } else {
          ((float*)Outv)[(size_t)mg * N + ng] = v;
        }
      }
}

// ---------------------------------------------------------------------------
// One attention head. Block = 16 queries of one (b,h). 8 waves.
// Q tile staged by TDM; K fragments register double-buffered across tiles;
// dual accumulators break the serial XDL chain; sched_group_barrier pins the
// load/WMMA overlap.  keepNeg is always valid (zeroed for head 0).
// ---------------------------------------------------------------------------
__global__ __launch_bounds__(256) void attn_head_kernel(
    const __bf16* __restrict__ Q, const __bf16* __restrict__ Kh,
    const __bf16* __restrict__ Vt, __bf16* __restrict__ ctx,
    float* __restrict__ attn_cum, const float* __restrict__ keepNeg, int h) {
  extern __shared__ unsigned char smem_raw[];
  float*  sS = (float*)smem_raw;                                  // 16 x 2048 f32
  __bf16* sP = (__bf16*)(smem_raw + (size_t)16 * S_ * sizeof(float));
  __bf16* sQ = sP;  // 8KB Q staging aliases sP (sP written much later)
  __shared__ float red[16][17];
  __shared__ float rowM[16], rowL[16];

  const int tid  = threadIdx.x;
  const int lane = tid & 31;
  const int wid  = tid >> 5;
  const int b    = blockIdx.y;
  const int q0   = blockIdx.x * 16;

  const __bf16* Qb = Q  + (((size_t)b * H_ + h) * S_ + q0) * DK_;
  const __bf16* Kb = Kh + (((size_t)b * H_ + h) * S_) * DK_;
  const __bf16* Vb = Vt + (((size_t)b * H_ + h) * DK_) * S_;  // [dk][s]
  const float scale = 0.0625f;  // 1/sqrt(DK)

  // -------- TDM: stage 16x256 bf16 Q tile (8KB contiguous) into LDS ------
  if (wid == 0) {
    tdm_load_2d(lds_offset(sQ), (unsigned long long)(size_t)Qb,
                /*tile_d0=*/1024, /*tile_d1=*/1, 1024ull, 1ull, 1024ull);
    __builtin_amdgcn_s_wait_tensorcnt(0);
  }
  __syncthreads();

  bf16x16 qF[8];
#pragma unroll
  for (int kk = 0; kk < 8; ++kk) qF[kk] = load_a_frag(sQ + kk * 32, DK_, lane);
  __syncthreads();  // everyone has qF before sP region is reused

  const f32x8 vzero = {};

  // -------- Phase 1: S = scale * Q K^T + mask --------------------------
  bf16x16 kA[8], kB[8];
#pragma unroll
  for (int kk = 0; kk < 8; ++kk)
    kA[kk] = load_b_frag(Kb + (size_t)(wid * 16) * DK_ + kk * 32, DK_, lane);

  for (int jt = wid; jt < S_ / 16; jt += 8) {
    const int jn = (jt + 8 < S_ / 16) ? jt + 8 : jt;  // clamped prefetch
#pragma unroll
    for (int kk = 0; kk < 8; ++kk)                     // next tile in flight
      kB[kk] = load_b_frag(Kb + (size_t)(jn * 16) * DK_ + kk * 32, DK_, lane);

    f32x8 a0 = vzero, a1 = vzero;                      // dual acc: chain/2
#pragma unroll
    for (int kk = 0; kk < 4; ++kk) a0 = wmma_bf16(qF[kk], kA[kk], a0);
#pragma unroll
    for (int kk = 4; kk < 8; ++kk) a1 = wmma_bf16(qF[kk], kA[kk], a1);
    const f32x8 acc = a0 + a1;

    const int key = jt * 16 + (lane & 15);
    const int mo  = (lane >> 4) << 3;
    const float kn = keepNeg[(size_t)b * S_ + key];    // branch-free (zeroed h0)
#pragma unroll
    for (int r = 0; r < 8; ++r)
      sS[(size_t)(mo + r) * S_ + key] = acc[r] * scale + kn;
#pragma unroll
    for (int kk = 0; kk < 8; ++kk) kA[kk] = kB[kk];

    __builtin_amdgcn_sched_group_barrier(0x020, 17, 0);  // K-tile + mask loads
    __builtin_amdgcn_sched_group_barrier(0x008, 8, 0);   // 8 WMMA
    __builtin_amdgcn_sched_group_barrier(0x200, 8, 0);   // score DS writes
  }
  __syncthreads();

  // -------- Phase 2: exact softmax, 16 threads per row -------------------
  const int rr = tid >> 4;
  const int cc = tid & 15;
  float mx = -3.0e38f;
  for (int k = cc; k < S_; k += 16) mx = fmaxf(mx, sS[(size_t)rr * S_ + k]);
  red[rr][cc] = mx;
  __syncthreads();
  if (cc == 0) {
    float m2 = red[rr][0];
#pragma unroll
    for (int t = 1; t < 16; ++t) m2 = fmaxf(m2, red[rr][t]);
    rowM[rr] = m2;
  }
  __syncthreads();
  const float rm = rowM[rr];
  float ls = 0.0f;
  for (int k = cc; k < S_; k += 16) ls += __expf(sS[(size_t)rr * S_ + k] - rm);
  red[rr][cc] = ls;
  __syncthreads();
  if (cc == 0) {
    float s2 = 0.0f;
#pragma unroll
    for (int t = 0; t < 16; ++t) s2 += red[rr][t];
    rowL[rr] = s2;
  }
  __syncthreads();
  const float rinv = 1.0f / rowL[rr];
  for (int k = cc; k < S_; k += 16)
    sP[(size_t)rr * S_ + k] =
        (__bf16)(__expf(sS[(size_t)rr * S_ + k] - rm) * rinv);
  __syncthreads();

  // -------- head_mean accumulation: column sums / S ----------------------
  for (int key = tid; key < S_; key += 256) {
    float cs = 0.0f;
#pragma unroll
    for (int m = 0; m < 16; ++m) cs += (float)sP[(size_t)m * S_ + key];
    atomicAdd(&attn_cum[(size_t)b * S_ + key], cs * (1.0f / (float)S_));
  }
  __syncthreads();

  // -------- Phase 3: O = P @ V, 2-wide unroll + dual accumulators --------
  for (int nt = wid * 2; nt < wid * 2 + 2; ++nt) {
    const int dk0 = nt * 16;
    f32x8 a0 = vzero, a1 = vzero;
    for (int ks = 0; ks < S_ / 32; ks += 2) {
      bf16x16 p0 = load_a_frag(sP + ks * 32, S_, lane);
      bf16x16 v0 = load_b_frag(Vb + (size_t)dk0 * S_ + ks * 32, S_, lane);
      bf16x16 p1 = load_a_frag(sP + (ks + 1) * 32, S_, lane);
      bf16x16 v1 = load_b_frag(Vb + (size_t)dk0 * S_ + (ks + 1) * 32, S_, lane);
      a0 = wmma_bf16(p0, v0, a0);
      a1 = wmma_bf16(p1, v1, a1);
      __builtin_amdgcn_sched_group_barrier(0x100, 4, 0);  // P frags (DS)
      __builtin_amdgcn_sched_group_barrier(0x020, 4, 0);  // V frags (VMEM)
      __builtin_amdgcn_sched_group_barrier(0x008, 2, 0);  // 2 WMMA
    }
    const f32x8 acc = a0 + a1;

    const int dk = dk0 + (lane & 15);
    const int mo = (lane >> 4) << 3;
#pragma unroll
    for (int r = 0; r < 8; ++r) {
      const int q = q0 + mo + r;
      ctx[((size_t)b * S_ + q) * D_ + h * DK_ + dk] = (__bf16)acc[r];
    }
  }
}

// ---------------------------------------------------------------------------
// Boost mask: z = lnorm(attn_cum) (ddof=1), prob = 0.4*sigmoid(50 z),
// keepNeg = bernoulli(prob) ? -1e9 : 0 (hash PRNG stands in for threefry).
// ---------------------------------------------------------------------------
__device__ __forceinline__ float hash_u01(unsigned x) {
  x ^= x >> 17; x *= 0xed5ad4bbu; x ^= x >> 11; x *= 0xac4c1b51u;
  x ^= x >> 15; x *= 0x31848babu; x ^= x >> 14;
  return (float)(x >> 8) * (1.0f / 16777216.0f);
}

__global__ __launch_bounds__(256) void boost_mask_kernel(
    const float* __restrict__ attn_cum, float* __restrict__ keepNeg,
    unsigned seed) {
  __shared__ float sred[256];
  const int b = blockIdx.x;
  const int tid = threadIdx.x;
  const float* row = attn_cum + (size_t)b * S_;

  float s = 0.0f;
  for (int k = tid; k < S_; k += 256) s += row[k];
  sred[tid] = s; __syncthreads();
  for (int off = 128; off > 0; off >>= 1) {
    if (tid < off) sred[tid] += sred[tid + off];
    __syncthreads();
  }
  const float mean = sred[0] * (1.0f / (float)S_);
  __syncthreads();

  float v = 0.0f;
  for (int k = tid; k < S_; k += 256) { float d = row[k] - mean; v += d * d; }
  sred[tid] = v; __syncthreads();
  for (int off = 128; off > 0; off >>= 1) {
    if (tid < off) sred[tid] += sred[tid + off];
    __syncthreads();
  }
  const float stdv = sqrtf(sred[0] / (float)(S_ - 1));  // unbiased (ddof=1)
  const float inv  = 1.0f / (stdv + 1e-7f);

  for (int k = tid; k < S_; k += 256) {
    float z = (row[k] - mean) * inv;
    float prob = 0.4f / (1.0f + __expf(-50.0f * z));
    float u = hash_u01(seed * 0x9E3779B9u +
                       (unsigned)(b * S_ + k) * 2654435761u + 0x85ebca6bu);
    keepNeg[(size_t)b * S_ + k] = (u < prob) ? NEG_ : 0.0f;
  }
}

// ---------------------------------------------------------------------------
extern "C" void kernel_launch(void* const* d_in, const int* in_sizes, int n_in,
                              void* d_out, int out_size, void* d_ws,
                              size_t ws_size, hipStream_t stream) {
  const float* query = (const float*)d_in[0];
  const float* key   = (const float*)d_in[1];
  const float* value = (const float*)d_in[2];
  const float* Wq = (const float*)d_in[3]; const float* bq = (const float*)d_in[4];
  const float* Wk = (const float*)d_in[5]; const float* bk = (const float*)d_in[6];
  const float* Wv = (const float*)d_in[7]; const float* bv = (const float*)d_in[8];
  const float* Wo = (const float*)d_in[9]; const float* bo = (const float*)d_in[10];

  const size_t BUF = (size_t)B_ * S_ * D_ * sizeof(__bf16);  // 32 MB each
  unsigned char* ws = (unsigned char*)d_ws;
  __bf16* Qw  = (__bf16*)(ws);
  __bf16* Kw  = (__bf16*)(ws + BUF);
  __bf16* Vtw = (__bf16*)(ws + 2 * BUF);
  __bf16* Ctx = (__bf16*)(ws + 3 * BUF);
  float* attn_cum = (float*)(ws + 4 * BUF);
  float* keepNeg  = (float*)(ws + 4 * BUF + (size_t)B_ * S_ * sizeof(float));

  hipMemsetAsync(attn_cum, 0, (size_t)B_ * S_ * sizeof(float), stream);
  hipMemsetAsync(keepNeg,  0, (size_t)B_ * S_ * sizeof(float), stream);  // head 0: no mask

  dim3 gGemm(128, 16), bGemm(256);
  gemm_wmma_kernel<0, false><<<gGemm, bGemm, 0, stream>>>(query, Wq, bq, Qw,  B_ * S_, D_, D_);
  gemm_wmma_kernel<0, false><<<gGemm, bGemm, 0, stream>>>(key,   Wk, bk, Kw,  B_ * S_, D_, D_);
  gemm_wmma_kernel<1, false><<<gGemm, bGemm, 0, stream>>>(value, Wv, bv, Vtw, B_ * S_, D_, D_);

  dim3 gAttn(S_ / 16, B_), bAttn(256);
  const size_t smem = (size_t)16 * S_ * (sizeof(float) + sizeof(__bf16));  // 192 KB
  for (int h = 0; h < H_; ++h) {
    attn_head_kernel<<<gAttn, bAttn, smem, stream>>>(Qw, Kw, Vtw, Ctx,
                                                     attn_cum, keepNeg, h);
    if (h < H_ - 1)
      boost_mask_kernel<<<dim3(B_), dim3(256), 0, stream>>>(attn_cum, keepNeg,
                                                            42u + (unsigned)h);
  }

  gemm_wmma_kernel<2, true><<<gGemm, bGemm, 0, stream>>>(Ctx, Wo, bo, d_out,
                                                         B_ * S_, D_, D_);
  // second tuple element: zeros((1,))
  hipMemsetAsync((float*)d_out + (size_t)B_ * S_ * D_, 0, sizeof(float), stream);
}
